// UniPhyModel_88579405513036
// MI455X (gfx1250) — matrix operand order repaired and checked by reading, not compiled, not run on hardware
//
#include <hip/hip_runtime.h>
#include <hip/hip_bf16.h>

// ---------------------------------------------------------------------------
// UniPhyModel forward for MI455X (gfx1250), wave32 + WMMA f32_16x16x32_f16.
// GEMM operands are pre-converted to f16 (A row-major, B packed [n][k]) so the
// hot loop is pure vector loads (b128) + v_wmma, no per-element cvt.
// ---------------------------------------------------------------------------

typedef __attribute__((ext_vector_type(16))) _Float16 v16h;
typedef __attribute__((ext_vector_type(8)))  _Float16 v8h;
typedef __attribute__((ext_vector_type(8)))  float    v8f;

#define NTOK 2048   // B*T*HP*WP tokens
#define TT   8      // T
#define SPA  256    // HP*WP
#define DD   256    // D
#define D2   512    // 2*D
#define FF   1024
#define NE   4
#define CONVK 4608  // 9*512

// ---------------- WMMA fragment helpers (A/B in contiguous k order) ----------------

// A fragment, 16x32 f16, A row-major MxK f16. lane 0-15: row m0+lane,
// K = k0+{0..15}; lane 16-31: same rows, K = k0+{16..31}.
__device__ __forceinline__ v16h make_a_frag(v8h lo, v8h hi) {
  v16h a;
#pragma unroll
  for (int e = 0; e < 8; ++e) { a[e] = lo[e]; a[8 + e] = hi[e]; }
  return a;
}

__device__ __forceinline__ v8f wmma_f16(v16h a, v16h b, v8f c) {
  return __builtin_amdgcn_wmma_f32_16x16x32_f16(false, a, false, b,
                                                (short)0, c, false, false);
}

__device__ __forceinline__ float gelu_f(float x) {
  float x3 = x * x * x;
  return 0.5f * x * (1.0f + tanhf(0.7978845608028654f * (x + 0.044715f * x3)));
}

__device__ __forceinline__ void store_tile(float* __restrict__ C, int N, int m0,
                                           int n, int lane, v8f acc, float bv,
                                           const float* __restrict__ rowscale,
                                           int rsStride, int accumulate, int act) {
#pragma unroll
  for (int r = 0; r < 8; ++r) {
    int m = m0 + r + ((lane >> 4) << 3);
    float v = acc[r] + bv;
    if (act == 1) v = gelu_f(v);
    if (rowscale) v *= rowscale[(size_t)m * rsStride];
    float* p = C + (size_t)m * N + n;
    if (accumulate) *p += v; else *p = v;
  }
}

// ---------------------------- generic WMMA GEMM -----------------------------
// C[M,N] (+)= act(A[M,K] @ B + bias) * rowscale.
// A: f16 row-major [m][k]. Bt: f16 packed [n][k] (K contiguous).
// Each wave computes a 16x32 tile (2 WMMAs per 32-deep K step).
__global__ __launch_bounds__(256) void gemm_wmma(
    const _Float16* __restrict__ A, const _Float16* __restrict__ Bt,
    float* __restrict__ C, int M, int N, int K,
    const float* __restrict__ bias,
    const float* __restrict__ rowscale, int rsStride,
    int accumulate, int act) {
  int wave = threadIdx.x >> 5;
  int lane = threadIdx.x & 31;
  int tile = blockIdx.x * (blockDim.x >> 5) + wave;
  int tn = N >> 5;                            // 32-wide N tiles
  int total = (M >> 4) * tn;
  if (tile >= total) return;                  // wave-uniform: EXEC stays all-1s
  int m0 = (tile / tn) << 4;
  int n0 = (tile % tn) << 5;

  const _Float16* arow  = A  + (size_t)(m0 + (lane & 15)) * K + ((lane >> 4) << 4);
  const _Float16* bcol0 = Bt + (size_t)(n0 + (lane & 15)) * K + ((lane >> 4) << 4);
  const _Float16* bcol1 = bcol0 + (size_t)16 * K;

  v8f acc0 = {}, acc1 = {};
  for (int k0 = 0; k0 < K; k0 += 32) {
    __builtin_prefetch(arow + k0 + 256, 0, 3);
    __builtin_prefetch(bcol0 + k0 + 256, 0, 3);
    v8h lo = *(const v8h*)(arow + k0);
    v8h hi = *(const v8h*)(arow + k0 + 8);
    v16h a  = make_a_frag(lo, hi);
    v16h b0 = *(const v16h*)(bcol0 + k0);
    v16h b1 = *(const v16h*)(bcol1 + k0);
    acc0 = wmma_f16(a, b0, acc0);
    acc1 = wmma_f16(a, b1, acc1);
  }
  int n = n0 + (lane & 15);
  float bv0 = bias ? bias[n] : 0.0f;
  float bv1 = bias ? bias[n + 16] : 0.0f;
  store_tile(C, N, m0, n,      lane, acc0, bv0, rowscale, rsStride, accumulate, act);
  store_tile(C, N, m0, n + 16, lane, acc1, bv1, rowscale, rsStride, accumulate, act);
}

// ------------------- 3x3 SAME conv 512->512 as implicit GEMM ----------------
// Zh: NTOK x 512 f16 (token-major NHWC). Bc: f16 packed [o][tap*512 + i].
__global__ __launch_bounds__(256) void conv3x3_wmma(
    const _Float16* __restrict__ Zh, const _Float16* __restrict__ Bc,
    const float* __restrict__ bias, float* __restrict__ Out) {
  int wave = threadIdx.x >> 5;
  int lane = threadIdx.x & 31;
  int tile = blockIdx.x * (blockDim.x >> 5) + wave;
  const int tn = D2 >> 5;                   // 16 tiles along N
  const int total = (NTOK >> 4) * tn;       // 2048 tiles
  if (tile >= total) return;
  int m0 = (tile / tn) << 4;
  int n0 = (tile % tn) << 5;

  int mloc = m0 + (lane & 15);
  int t  = mloc >> 8;
  int hp = (mloc >> 4) & 15;
  int wp = mloc & 15;
  int aoff = (lane >> 4) << 4;

  const _Float16* bcol0 = Bc + (size_t)(n0 + (lane & 15)) * CONVK + ((lane >> 4) << 4);
  const _Float16* bcol1 = bcol0 + (size_t)16 * CONVK;

  v8f acc0 = {}, acc1 = {};
  for (int tap = 0; tap < 9; ++tap) {
    int dy = tap / 3 - 1, dx = tap % 3 - 1;
    int hp2 = hp + dy, wp2 = wp + dx;
    bool valid = ((unsigned)hp2 < 16u) && ((unsigned)wp2 < 16u);
    const _Float16* row = Zh + (size_t)((t << 8) + (hp2 << 4) + wp2) * D2 + aoff;
    int kb = tap * D2;
    for (int k0 = 0; k0 < D2; k0 += 32) {
      v8h lo = {}, hi = {};
      if (valid) {
        lo = *(const v8h*)(row + k0);
        hi = *(const v8h*)(row + k0 + 8);
      }
      v16h a  = make_a_frag(lo, hi);
      v16h b0 = *(const v16h*)(bcol0 + kb + k0);
      v16h b1 = *(const v16h*)(bcol1 + kb + k0);
      acc0 = wmma_f16(a, b0, acc0);
      acc1 = wmma_f16(a, b1, acc1);
    }
  }
  int n = n0 + (lane & 15);
  store_tile(Out, D2, m0, n,      lane, acc0, bias[n],      nullptr, 0, 0, 0);
  store_tile(Out, D2, m0, n + 16, lane, acc1, bias[n + 16], nullptr, 0, 0, 0);
}

// --------------------------- pack / convert kernels -------------------------

__global__ void cvt_f16_k(const float* __restrict__ in, _Float16* __restrict__ out, int n) {
  int i = blockIdx.x * blockDim.x + threadIdx.x;
  if (i < n) out[i] = (_Float16)in[i];
}

// Bt[n*K + k] = B[k*bsK + n*bsN] as f16
__global__ void pack_b_f16_k(const float* __restrict__ B, _Float16* __restrict__ Bt,
                             int K, int N, long long bsK, long long bsN) {
  int i = blockIdx.x * blockDim.x + threadIdx.x;
  if (i >= N * K) return;
  int nn = i / K, k = i % K;
  Bt[i] = (_Float16)B[(long long)k * bsK + (long long)nn * bsN];
}

// Bc[o*4608 + tap*512 + i] = conv_w[o][i][tap]  (conv_w: O,I,3,3)
__global__ void pack_conv_f16_k(const float* __restrict__ Wc, _Float16* __restrict__ Bc) {
  int i = blockIdx.x * blockDim.x + threadIdx.x;
  if (i >= D2 * CONVK) return;
  int o = i / CONVK, r = i % CONVK;
  int tap = r / D2, ii = r % D2;
  Bc[i] = (_Float16)Wc[(size_t)o * CONVK + (size_t)ii * 9 + tap];
}

// complex D x D -> packed f16 [n][k] of real 2D x 2D [[Re,Im],[-Im,Re]]
__global__ void build_cmat_f16_k(const float* __restrict__ Re, const float* __restrict__ Im,
                                 _Float16* __restrict__ Bt) {
  int i = blockIdx.x * blockDim.x + threadIdx.x;
  if (i >= D2 * D2) return;
  int n = i / D2, k = i % D2;
  float v;
  if (k < DD) v = (n < DD) ? Re[k * DD + n] : Im[k * DD + (n - DD)];
  else        v = (n < DD) ? -Im[(k - DD) * DD + n] : Re[(k - DD) * DD + (n - DD)];
  Bt[i] = (_Float16)v;
}

// ------------------------------ small kernels -------------------------------

__global__ void fill_zero(float* p, int n) {
  int i = blockIdx.x * blockDim.x + threadIdx.x;
  if (i < n) p[i] = 0.0f;
}

// x (1,8,4,256,256) -> xp (2048, 1024) ; feature = c*256 + py*16 + px
__global__ void patchify_k(const float* __restrict__ x, float* __restrict__ xp) {
  int i = blockIdx.x * blockDim.x + threadIdx.x;
  if (i >= NTOK * 1024) return;
  int n = i >> 10, f = i & 1023;
  int c = f >> 8, py = (f >> 4) & 15, px = f & 15;
  int t = n >> 8, hp = (n >> 4) & 15, wp = n & 15;
  xp[i] = x[(((size_t)(t * 4 + c) * 256) + hp * 16 + py) * 256 + wp * 16 + px];
}

__global__ void unpatchify_k(const float* __restrict__ yo, float* __restrict__ out) {
  int i = blockIdx.x * blockDim.x + threadIdx.x;
  if (i >= NTOK * 1024) return;
  int n = i >> 10, f = i & 1023;
  int c = f >> 8, py = (f >> 4) & 15, px = f & 15;
  int t = n >> 8, hp = (n >> 4) & 15, wp = n & 15;
  out[(((size_t)(t * 4 + c) * 256) + hp * 16 + py) * 256 + wp * 16 + px] = yo[i];
}

// per-token LayerNorm over C channels (LDS tree reduction)
__global__ __launch_bounds__(256) void ln_k(const float* __restrict__ X,
                                            float* __restrict__ Out,
                                            const float* __restrict__ w,
                                            const float* __restrict__ b, int C) {
  __shared__ float red[256];
  int tok = blockIdx.x, tid = threadIdx.x;
  const float* xr = X + (size_t)tok * C;
  float s = 0.0f;
  for (int j = tid; j < C; j += 256) s += xr[j];
  red[tid] = s; __syncthreads();
  for (int st = 128; st > 0; st >>= 1) {
    if (tid < st) red[tid] += red[tid + st];
    __syncthreads();
  }
  float mean = red[0] / (float)C;
  __syncthreads();
  float v = 0.0f;
  for (int j = tid; j < C; j += 256) { float d = xr[j] - mean; v += d * d; }
  red[tid] = v; __syncthreads();
  for (int st = 128; st > 0; st >>= 1) {
    if (tid < st) red[tid] += red[tid + st];
    __syncthreads();
  }
  float inv = rsqrtf(red[0] / (float)C + 1e-5f);
  for (int j = tid; j < C; j += 256)
    Out[(size_t)tok * C + j] = (xr[j] - mean) * inv * w[j] + b[j];
}

__global__ void add2_k(const float* a, const float* b, float* o, int n) {
  int i = blockIdx.x * blockDim.x + threadIdx.x;
  if (i < n) o[i] = a[i] + b[i];
}

__global__ void add3_k(const float* a, const float* b, const float* c, float* o, int n) {
  int i = blockIdx.x * blockDim.x + threadIdx.x;
  if (i < n) o[i] = a[i] + b[i] + c[i];
}

// spatial mean: xm[t][col] = mean_s XE[(t*256+s)*512 + col]
__global__ void mean_sp_k(const float* __restrict__ XE, float* __restrict__ xm) {
  int i = blockIdx.x * blockDim.x + threadIdx.x;
  if (i >= TT * D2) return;
  int t = i / D2, col = i % D2;
  float s = 0.0f;
  const float* base = XE + (size_t)t * SPA * D2 + col;
  for (int sp = 0; sp < SPA; ++sp) s += base[(size_t)sp * D2];
  xm[i] = s * (1.0f / SPA);
}

// source[t][e] = sum_d xm_c[t][d] * Ws_c[d][e]
__global__ void source_k(const float* __restrict__ xm, const float* __restrict__ Wsr,
                         const float* __restrict__ Wsi, float* __restrict__ src) {
  int i = blockIdx.x * blockDim.x + threadIdx.x;
  if (i >= TT * DD) return;
  int t = i / DD, e = i % DD;
  float sr = 0.0f, si = 0.0f;
  const float* xmr = xm + (size_t)t * D2;
  for (int d = 0; d < DD; ++d) {
    float xr = xmr[d], xi = xmr[DD + d];
    float wr = Wsr[d * DD + e], wi = Wsi[d * DD + e];
    sr += xr * wr - xi * wi;
    si += xr * wi + xi * wr;
  }
  src[(size_t)t * D2 + e] = sr;
  src[(size_t)t * D2 + DD + e] = si;
}

// gate[t][d] = sigmoid(feat[t] . Wg[:,d] + bg[d])
__global__ void gate_k(const float* __restrict__ xm, const float* __restrict__ Wg,
                       const float* __restrict__ bg, float* __restrict__ gate) {
  int i = blockIdx.x * blockDim.x + threadIdx.x;
  if (i >= TT * DD) return;
  int t = i / DD, d = i % DD;
  float s = bg[d];
  const float* xmr = xm + (size_t)t * D2;
  for (int j = 0; j < D2; ++j) s += xmr[j] * Wg[(size_t)j * DD + d];
  gate[i] = 1.0f / (1.0f + expf(-s));
}

// sequential flux scan over T (updates flux in place)
__global__ void fluxscan_k(const float* __restrict__ gate, const float* __restrict__ src,
                           float* __restrict__ flux) {
  int d = blockIdx.x * blockDim.x + threadIdx.x;
  if (d >= DD) return;
  float fr = flux[d], fi = flux[DD + d];
  for (int t = 0; t < TT; ++t) {
    float g = gate[t * DD + d];
    float sr = src[(size_t)t * D2 + d], si = src[(size_t)t * D2 + DD + d];
    fr = g * fr + (1.0f - g) * sr;
    fi = g * fi + (1.0f - g) * si;
  }
  flux[d] = fr; flux[DD + d] = fi;
}

// op_decay a = exp(lam*dt), op_forcing = (a-1)/lam, lam = -(softplus(lr)+0.01)+i*li
__global__ void decay_k(const float* __restrict__ lam_re, const float* __restrict__ lam_im,
                        const float* __restrict__ dt, float* __restrict__ dbuf) {
  int i = blockIdx.x * blockDim.x + threadIdx.x;
  if (i >= TT * DD) return;
  int t = i / DD, e = i % DD;
  float lrx = lam_re[e];
  float sp = (lrx > 20.0f) ? lrx : log1pf(expf(lrx));
  float lr = -(sp + 0.01f);
  float li = lam_im[e];
  float dtv = dt[t];
  float mag = expf(lr * dtv);
  float ar = mag * cosf(li * dtv);
  float ai = mag * sinf(li * dtv);
  float den = lr * lr + li * li;
  float nr = ar - 1.0f, ni = ai;
  float pr = (nr * lr + ni * li) / den;
  float pi = (ni * lr - nr * li) / den;
  float* a_re = dbuf, *a_im = dbuf + TT * DD;
  float* p_re = dbuf + 2 * TT * DD, *p_im = dbuf + 3 * TT * DD;
  a_re[i] = ar; a_im[i] = ai; p_re[i] = pr; p_im[i] = pi;
}

// forcing (in place on XE): F = XE*g + src*(1-g)
__global__ void forcing_k(float* __restrict__ XE, const float* __restrict__ gate,
                          const float* __restrict__ src) {
  int i = blockIdx.x * blockDim.x + threadIdx.x;
  if (i >= NTOK * DD) return;
  int n = i / DD, e = i % DD;
  int t = n >> 8;
  float g = gate[t * DD + e];
  float sr = src[(size_t)t * D2 + e], si = src[(size_t)t * D2 + DD + e];
  size_t re = (size_t)n * D2 + e, im = re + DD;
  XE[re] = XE[re] * g + sr * (1.0f - g);
  XE[im] = XE[im] * g + si * (1.0f - g);
}

// h scan over T: hs[t] = a[t]*hs[t-1] + F[t]*opf[t] ; out = hs + h_prev*a[t]
__global__ void hscan_k(const float* __restrict__ F, const float* __restrict__ dbuf,
                        float* __restrict__ hprev, float* __restrict__ H) {
  int i = blockIdx.x * blockDim.x + threadIdx.x;
  if (i >= SPA * DD) return;
  int s = i / DD, e = i % DD;
  const float* a_re = dbuf, *a_im = dbuf + TT * DD;
  const float* p_re = dbuf + 2 * TT * DD, *p_im = dbuf + 3 * TT * DD;
  float hpr = hprev[(size_t)s * D2 + e], hpi = hprev[(size_t)s * D2 + DD + e];
  float hr = 0.0f, hi = 0.0f, outr = 0.0f, outi = 0.0f;
  for (int t = 0; t < TT; ++t) {
    int te = t * DD + e;
    float ar = a_re[te], ai = a_im[te], pr = p_re[te], pi = p_im[te];
    int n = t * SPA + s;
    float fr = F[(size_t)n * D2 + e], fi = F[(size_t)n * D2 + DD + e];
    float ur = fr * pr - fi * pi, ui = fr * pi + fi * pr;
    float nhr = ar * hr - ai * hi + ur;
    float nhi = ar * hi + ai * hr + ui;
    hr = nhr; hi = nhi;
    outr = hr + (hpr * ar - hpi * ai);
    outi = hi + (hpr * ai + hpi * ar);
    H[(size_t)n * D2 + e] = outr;
    H[(size_t)n * D2 + DD + e] = outi;
  }
  hprev[(size_t)s * D2 + e] = outr;
  hprev[(size_t)s * D2 + DD + e] = outi;
}

// router: probs = softmax(tok @ Wr) per token
__global__ void router_k(const float* __restrict__ TOK, const float* __restrict__ Wr,
                         float* __restrict__ probs) {
  int n = blockIdx.x * blockDim.x + threadIdx.x;
  if (n >= NTOK) return;
  const float* trow = TOK + (size_t)n * D2;
  float l0 = 0, l1 = 0, l2 = 0, l3 = 0;
  for (int j = 0; j < D2; ++j) {
    float tv = trow[j];
    const float* w = Wr + (size_t)j * NE;
    l0 += tv * w[0]; l1 += tv * w[1]; l2 += tv * w[2]; l3 += tv * w[3];
  }
  float m = fmaxf(fmaxf(l0, l1), fmaxf(l2, l3));
  float e0 = expf(l0 - m), e1 = expf(l1 - m), e2 = expf(l2 - m), e3 = expf(l3 - m);
  float inv = 1.0f / (e0 + e1 + e2 + e3);
  float* p = probs + (size_t)n * NE;
  p[0] = e0 * inv; p[1] = e1 * inv; p[2] = e2 * inv; p[3] = e3 * inv;
}

// ------------------------------- launcher -----------------------------------

static inline void gemm_launch(hipStream_t s, const _Float16* A, const _Float16* Bt,
                               float* C, int M, int N, int K,
                               const float* bias, const float* rs, int rsStride,
                               int acc, int act) {
  int tiles = (M / 16) * (N / 32);
  int blocks = (tiles + 7) / 8;
  gemm_wmma<<<blocks, 256, 0, s>>>(A, Bt, C, M, N, K, bias, rs, rsStride, acc, act);
}

static inline void cvt_launch(hipStream_t s, const float* in, _Float16* out, int n) {
  cvt_f16_k<<<(n + 255) / 256, 256, 0, s>>>(in, out, n);
}

extern "C" void kernel_launch(void* const* d_in, const int* in_sizes, int n_in,
                              void* d_out, int out_size, void* d_ws, size_t ws_size,
                              hipStream_t stream) {
  const float* x      = (const float*)d_in[0];
  const float* dt     = (const float*)d_in[1];
  const float* Wenc   = (const float*)d_in[2];
  const float* benc   = (const float*)d_in[3];
  const float* Wdec   = (const float*)d_in[4];
  const float* bdec   = (const float*)d_in[5];
  const float* ln_sp_w= (const float*)d_in[6];
  const float* ln_sp_b= (const float*)d_in[7];
  const float* conv_w = (const float*)d_in[8];
  const float* conv_b = (const float*)d_in[9];
  const float* E_re   = (const float*)d_in[10];
  const float* E_im   = (const float*)d_in[11];
  const float* Ed_re  = (const float*)d_in[12];
  const float* Ed_im  = (const float*)d_in[13];
  const float* Ws_re  = (const float*)d_in[14];
  const float* Ws_im  = (const float*)d_in[15];
  const float* Wg     = (const float*)d_in[16];
  const float* bg     = (const float*)d_in[17];
  const float* lam_re = (const float*)d_in[18];
  const float* lam_im = (const float*)d_in[19];
  const float* ln_t_w = (const float*)d_in[20];
  const float* ln_t_b = (const float*)d_in[21];
  const float* Wr     = (const float*)d_in[22];
  const float* W1     = (const float*)d_in[23];
  const float* W2     = (const float*)d_in[24];
  float* out = (float*)d_out;

  // workspace partition (float units; every size is a multiple of 512 floats,
  // so all f16 sub-buffers stay >=2KB aligned for v16h loads)
  float* ws = (float*)d_ws;
  size_t off = 0;
  auto alloc = [&](size_t n) { float* p = ws + off; off += n; return p; };
  float* R0   = alloc((size_t)NTOK * 1024);   // xp / per-expert hidden / decoder out
  float* Z    = alloc((size_t)NTOK * D2);
  float* Zn   = alloc((size_t)NTOK * D2);
  float* Zs   = alloc((size_t)NTOK * D2);
  float* X    = alloc((size_t)NTOK * D2);
  float* XE   = alloc((size_t)NTOK * D2);     // x_eigen, then forcing in place
  float* H    = alloc((size_t)NTOK * D2);
  float* XO   = alloc((size_t)NTOK * D2);
  float* TOK  = alloc((size_t)NTOK * D2);
  float* Y    = alloc((size_t)NTOK * D2);
  _Float16* Ah  = (_Float16*)alloc((size_t)NTOK * 1024 / 2);  // A staging (f16)
  _Float16* Ah2 = (_Float16*)alloc((size_t)NTOK * 1024 / 2);  // hid staging (f16)
  _Float16* Bh  = (_Float16*)alloc((size_t)D2 * CONVK / 2);   // B staging (f16)
  float* xm   = alloc((size_t)TT * D2);
  float* src  = alloc((size_t)TT * D2);
  float* gate = alloc((size_t)TT * DD);
  float* dbuf = alloc((size_t)4 * TT * DD);
  float* hprev= alloc((size_t)SPA * D2);
  float* flux = alloc((size_t)D2);
  float* probs= alloc((size_t)NTOK * NE);
  (void)ws_size; (void)in_sizes; (void)n_in; (void)out_size;

  const int EW = NTOK * D2;

  // init recurrent state
  fill_zero<<<(SPA * D2 + 255) / 256, 256, 0, stream>>>(hprev, SPA * D2);
  fill_zero<<<1, 256, 0, stream>>>(flux, D2);

  // ---------------- encoder ----------------
  patchify_k<<<(NTOK * 1024 + 255) / 256, 256, 0, stream>>>(x, R0);
  cvt_launch(stream, R0, Ah, NTOK * 1024);
  pack_b_f16_k<<<(D2 * 1024 + 255) / 256, 256, 0, stream>>>(Wenc, Bh, 1024, D2, D2, 1);
  gemm_launch(stream, Ah, Bh, Z, NTOK, D2, 1024, benc, nullptr, 0, 0, 0);

  for (int l = 0; l < 2; ++l) {
    const float* lnw  = ln_sp_w + l * D2;
    const float* lnb  = ln_sp_b + l * D2;
    const float* cw   = conv_w + (size_t)l * D2 * CONVK;
    const float* cb   = conv_b + l * D2;
    const float* Er   = E_re  + (size_t)l * DD * DD;
    const float* Ei   = E_im  + (size_t)l * DD * DD;
    const float* Edr  = Ed_re + (size_t)l * DD * DD;
    const float* Edi  = Ed_im + (size_t)l * DD * DD;
    const float* Wsr  = Ws_re + (size_t)l * DD * DD;
    const float* Wsi  = Ws_im + (size_t)l * DD * DD;
    const float* Wgl  = Wg + (size_t)l * D2 * DD;
    const float* bgl  = bg + l * DD;
    const float* lrl  = lam_re + l * DD;
    const float* lil  = lam_im + l * DD;
    const float* ltw  = ln_t_w + l * D2;
    const float* ltb  = ln_t_b + l * D2;
    const float* Wrl  = Wr + (size_t)l * D2 * NE;
    const float* W1l  = W1 + (size_t)l * NE * D2 * FF;
    const float* W2l  = W2 + (size_t)l * NE * FF * D2;

    // spatial LN + conv residual
    ln_k<<<NTOK, 256, 0, stream>>>(Z, Zn, lnw, lnb, D2);
    cvt_launch(stream, Zn, Ah, EW);
    pack_conv_f16_k<<<(D2 * CONVK + 255) / 256, 256, 0, stream>>>(cw, Bh);
    conv3x3_wmma<<<(2048 + 7) / 8, 256, 0, stream>>>(Ah, Bh, cb, Zs);
    add2_k<<<(EW + 255) / 256, 256, 0, stream>>>(Z, Zs, X, EW);

    // eigen projection: XE = X @ [[Er,Ei],[-Ei,Er]]
    build_cmat_f16_k<<<(D2 * D2 + 255) / 256, 256, 0, stream>>>(Er, Ei, Bh);
    cvt_launch(stream, X, Ah, EW);
    gemm_launch(stream, Ah, Bh, XE, NTOK, D2, D2, nullptr, nullptr, 0, 0, 0);

    // temporal gating scan (tiny)
    mean_sp_k<<<(TT * D2 + 255) / 256, 256, 0, stream>>>(XE, xm);
    source_k<<<(TT * DD + 255) / 256, 256, 0, stream>>>(xm, Wsr, Wsi, src);
    gate_k<<<(TT * DD + 255) / 256, 256, 0, stream>>>(xm, Wgl, bgl, gate);
    fluxscan_k<<<1, 256, 0, stream>>>(gate, src, flux);
    decay_k<<<(TT * DD + 255) / 256, 256, 0, stream>>>(lrl, lil, dt, dbuf);
    forcing_k<<<(NTOK * DD + 255) / 256, 256, 0, stream>>>(XE, gate, src);
    hscan_k<<<(SPA * DD + 255) / 256, 256, 0, stream>>>(XE, dbuf, hprev, H);

    // back-projection: XO = H @ [[Edr,Edi],[-Edi,Edr]]
    build_cmat_f16_k<<<(D2 * D2 + 255) / 256, 256, 0, stream>>>(Edr, Edi, Bh);
    cvt_launch(stream, H, Ah, EW);
    gemm_launch(stream, Ah, Bh, XO, NTOK, D2, D2, nullptr, nullptr, 0, 0, 0);

    // temporal LN + router + MoE FFN
    ln_k<<<NTOK, 256, 0, stream>>>(XO, TOK, ltw, ltb, D2);
    router_k<<<(NTOK + 255) / 256, 256, 0, stream>>>(TOK, Wrl, probs);
    cvt_launch(stream, TOK, Ah, EW);                  // TOK f16 reused by all experts
    for (int e = 0; e < NE; ++e) {
      const float* W1e = W1l + (size_t)e * D2 * FF;
      const float* W2e = W2l + (size_t)e * FF * D2;
      pack_b_f16_k<<<(FF * D2 + 255) / 256, 256, 0, stream>>>(W1e, Bh, D2, FF, FF, 1);
      gemm_launch(stream, Ah, Bh, R0, NTOK, FF, D2, nullptr, nullptr, 0, 0, /*gelu*/1);
      cvt_launch(stream, R0, Ah2, NTOK * FF);
      pack_b_f16_k<<<(FF * D2 + 255) / 256, 256, 0, stream>>>(W2e, Bh, FF, D2, D2, 1);
      gemm_launch(stream, Ah2, Bh, Y, NTOK, D2, FF, nullptr, probs + e, NE,
                  /*acc=*/(e > 0), 0);
    }

    // residual: z_next = x + (xn + delta)
    add3_k<<<(EW + 255) / 256, 256, 0, stream>>>(X, TOK, Y, Z, EW);
  }

  // ---------------- decoder ----------------
  cvt_launch(stream, Z, Ah, EW);
  pack_b_f16_k<<<(1024 * D2 + 255) / 256, 256, 0, stream>>>(Wdec, Bh, D2, 1024, 1024, 1);
  gemm_launch(stream, Ah, Bh, R0, NTOK, 1024, D2, bdec, nullptr, 0, 0, 0);
  unpatchify_k<<<(NTOK * 1024 + 255) / 256, 256, 0, stream>>>(R0, out);
}
